// LocalMultiScale_44590350467263
// MI455X (gfx1250) — compile-verified, matrix operand order checked by reading, not run on hardware
//
#include <hip/hip_runtime.h>
#include <hip/hip_bf16.h>
#include <cstdint>

// ---------------------------------------------------------------------------
// CDNA5 (gfx1250) wave32 WMMA types
// ---------------------------------------------------------------------------
typedef __attribute__((ext_vector_type(16))) __bf16 v16bf;
typedef __attribute__((ext_vector_type(8)))  float  v8f;

union BFrag { uint32_t u[8]; v16bf v; };

__device__ inline uint16_t f2bf(float x) {
  uint32_t a = __builtin_bit_cast(uint32_t, x);
  a += 0x7FFFu + ((a >> 16) & 1u);          // round-to-nearest-even
  return (uint16_t)(a >> 16);
}
__device__ inline uint32_t pack2bf(float lo, float hi) {
  uint32_t a = __builtin_bit_cast(uint32_t, lo);
  uint32_t b = __builtin_bit_cast(uint32_t, hi);
  a += 0x7FFFu + ((a >> 16) & 1u);
  b += 0x7FFFu + ((b >> 16) & 1u);
  return (a >> 16) | (b & 0xFFFF0000u);
}

// ---------------------------------------------------------------------------
// Weight packing: fp32 [Co][Ci][kh][kw] -> bf16 [Co][K], K = ks*ks*Ci,
// K order = tap*Ci + ci (tap = kh*ks + kw) so a 32-wide K step shares one tap.
// ---------------------------------------------------------------------------
__global__ void pack_w3x3(const float* __restrict__ src, uint16_t* __restrict__ dst,
                          int Co, int Ci) {
  int K = 9 * Ci, total = Co * K;
  for (int idx = blockIdx.x * blockDim.x + threadIdx.x; idx < total;
       idx += gridDim.x * blockDim.x) {
    int co = idx / K, k = idx - co * K;
    int tap = k / Ci, ci = k - tap * Ci;
    dst[idx] = f2bf(src[(size_t)(co * Ci + ci) * 9 + tap]);
  }
}
__global__ void pack_w1x1(const float* __restrict__ src, uint16_t* __restrict__ dst, int n) {
  for (int i = blockIdx.x * blockDim.x + threadIdx.x; i < n; i += gridDim.x * blockDim.x)
    dst[i] = f2bf(src[i]);
}
__global__ void concat35(const float* a25, const float* a9, const float* a1, float* dst) {
  int t = threadIdx.x;
  if (t < 25) dst[t] = a25[t];
  else if (t < 34) dst[t] = a9[t - 25];
  else if (t == 34) dst[t] = a1[0];
}

// ---------------------------------------------------------------------------
// Implicit-GEMM conv, bf16 WMMA 16x16x32, fp32 accumulate. All geometry is
// compile-time so the im2col address math strength-reduces to shifts/mads.
// Block = 128 threads (4 wave32), tile = 64 Co x 64 output pixels.
// Thread t permanently owns LDS column n = t&63 and k-pairs kp = (t>>6)+2i,
// so pixel geometry is hoisted out of the K loop. LDS B-tile double-buffered:
// one barrier per K-step; next step's loads overlap current step's WMMAs.
// ---------------------------------------------------------------------------
template <int KS, int STRIDE, int PAD, int CI, int HI, int WI, int HO, int WO>
__global__ __launch_bounds__(128) void conv_wmma(
    const float* __restrict__ in, long long inStrideC, long long inStrideB,
    const float* __restrict__ ascale, const float* __restrict__ ashift, int reluIn,
    const uint16_t* __restrict__ wp, const float* __restrict__ bias,
    float* __restrict__ out, int Co, int Bn) {
  constexpr int K   = KS * KS * CI;   // CI % 32 == 0 -> one tap per 32-wide K step
  constexpr int HWo = HO * WO;
  const int Nout   = Bn * HWo;
  const int nBase  = blockIdx.x * 64;
  const int coBase = blockIdx.y * 64;
  const int t    = threadIdx.x;
  const int wave = t >> 5, lane = t & 31;
  const int lm = lane & 15, lh = lane >> 4;

  __shared__ uint32_t ldsB[2][16 * 64];

  // ---- loop-invariant pixel geometry for this thread's loader role ----
  const int n  = t & 63;
  const int ng = nBase + n;
  const int b  = ng / HWo;            // constant divisor -> shifts
  const int hw = ng - b * HWo;
  const int oh = hw / WO, ow = hw - oh * WO;
  const size_t pixBase = (size_t)b * inStrideB;
  const int ih0 = oh * STRIDE - PAD, iw0 = ow * STRIDE - PAD;

  v8f acc0 = {}, acc1 = {}, acc2 = {}, acc3 = {};
  const uint32_t* wp32 = (const uint32_t*)wp;
  int rowA = coBase + wave * 16 + lm;
  if (rowA >= Co) rowA = Co - 1;       // clamp loads, mask stores later

  auto loadStep = [&](int k0, int buf) {
    const int tap = k0 / CI;           // constant divisor
    const int ci0 = k0 - tap * CI;
    const int kh = tap / KS, kw = tap - kh * KS;
    const int ih = ih0 + kh, iw = iw0 + kw;
    const bool valid = (ih >= 0) & (ih < HI) & (iw >= 0) & (iw < WI);
    const size_t sbase = pixBase + (size_t)ih * WI + iw;
    #pragma unroll
    for (int i = 0; i < 8; ++i) {
      int kp = (t >> 6) + 2 * i;       // 0,2,..14 or 1,3,..15
      int ci = ci0 + kp * 2;
      float v0 = 0.f, v1 = 0.f;
      if (valid) {
        v0 = in[(size_t)ci * inStrideC + sbase];
        v1 = in[(size_t)(ci + 1) * inStrideC + sbase];
        if (ascale) {
          v0 = v0 * ascale[ci]     + ashift[ci];
          v1 = v1 * ascale[ci + 1] + ashift[ci + 1];
        }
        if (reluIn) { v0 = fmaxf(v0, 0.f); v1 = fmaxf(v1, 0.f); }
      }
      ldsB[buf][kp * 64 + n] = pack2bf(v0, v1);
    }
  };

  loadStep(0, 0);
  for (int k0 = 0; k0 < K; k0 += 32) {
    __syncthreads();
    const int buf = (k0 >> 5) & 1;
    if (k0 + 32 < K) loadStep(k0 + 32, buf ^ 1);   // overlaps WMMAs below

    // ---- A fragment: 16x32 bf16 of packed weights (per-lane gather) ----
    BFrag aF;
    const size_t abase = (size_t)rowA * K + k0;
    #pragma unroll
    for (int v = 0; v < 8; ++v) {
      int kb = (v >> 2) * 16 + lh * 8 + (v & 3) * 2;
      aF.u[v] = wp32[(abase + kb) >> 1];
    }
    if (k0 + 64 < K) __builtin_prefetch(&wp[abase + 64], 0, 1);  // global_prefetch_b8

    // ---- 4 WMMAs: A reused across four 16-col N subtiles ----
    #pragma unroll
    for (int j = 0; j < 4; ++j) {
      BFrag bF;
      int nn = j * 16 + lm;
      #pragma unroll
      for (int v = 0; v < 8; ++v) {
        int kp = (v >> 2) * 8 + lh * 4 + (v & 3);
        bF.u[v] = ldsB[buf][kp * 64 + nn];
      }
      v8f& acc = (j == 0 ? acc0 : j == 1 ? acc1 : j == 2 ? acc2 : acc3);
      acc = __builtin_amdgcn_wmma_f32_16x16x32_bf16(
          false, aF.v, false, bF.v, (short)0, acc, false, false);
    }
  }

  // ---- store raw fp32 (+bias); C/D layout: lane=(m>=8)*16+n, vgpr r = m%8 ----
  const int co0 = coBase + wave * 16;
  #pragma unroll
  for (int j = 0; j < 4; ++j) {
    v8f acc = (j == 0 ? acc0 : j == 1 ? acc1 : j == 2 ? acc2 : acc3);
    int nn = nBase + j * 16 + lm;
    #pragma unroll
    for (int r = 0; r < 8; ++r) {
      int co = co0 + lh * 8 + r;
      if (co < Co) out[(size_t)co * Nout + nn] = acc[r] + bias[co];
    }
  }
}

// ---------------------------------------------------------------------------
// Per-channel batch stats: one block per channel, deterministic tree reduce.
// ---------------------------------------------------------------------------
__global__ __launch_bounds__(256) void bn_stats(const float* __restrict__ x,
                                                float* __restrict__ sum,
                                                float* __restrict__ sumsq, int N) {
  int c = blockIdx.x;
  const float* row = x + (size_t)c * N;
  float s = 0.f, ss = 0.f;
  for (int i = threadIdx.x; i < N; i += 256) { float v = row[i]; s += v; ss += v * v; }
  __shared__ float shs[256], shq[256];
  shs[threadIdx.x] = s; shq[threadIdx.x] = ss;
  __syncthreads();
  for (int o = 128; o > 0; o >>= 1) {
    if (threadIdx.x < o) { shs[threadIdx.x] += shs[threadIdx.x + o];
                           shq[threadIdx.x] += shq[threadIdx.x + o]; }
    __syncthreads();
  }
  if (threadIdx.x == 0) { sum[c] = shs[0]; sumsq[c] = shq[0]; }
}

__global__ void bn_finalize(const float* sum, const float* sumsq,
                            const float* g, const float* be,
                            float* scale, float* shift, int C, float invN) {
  int c = blockIdx.x * blockDim.x + threadIdx.x;
  if (c < C) {
    float m = sum[c] * invN;
    float v = sumsq[c] * invN - m * m;
    float s = g[c] * rsqrtf(v + 1e-5f);
    scale[c] = s;
    shift[c] = be[c] - m * s;
  }
}

// x34 = relu(bn2(d1) + bn3(d2)), fp32 matrix [256][N]
__global__ void x34_fuse(const float* __restrict__ d1, const float* __restrict__ d2,
                         const float* s2, const float* h2,
                         const float* s3, const float* h3,
                         float* __restrict__ out, int N) {
  int total = 256 * N;
  for (int i = blockIdx.x * blockDim.x + threadIdx.x; i < total;
       i += gridDim.x * blockDim.x) {
    int c = i / N;
    float v = d1[i] * s2[c] + h2[c] + d2[i] * s3[c] + h3[c];
    out[i] = fmaxf(v, 0.f);
  }
}

// ---------------------------------------------------------------------------
// Dynamic filtering: softmax(25), softmax(9), pointwise gate; writes the
// 3C concat directly in GEMM matrix layout [768][N]. 16 pixels/block.
// ---------------------------------------------------------------------------
__global__ __launch_bounds__(256) void dynfilter(
    const float* __restrict__ sraw, const float* __restrict__ sc,
    const float* __restrict__ sh, const float* __restrict__ x5,
    float* __restrict__ xc, int Bn) {
  const int HW = 4096, N = Bn * HW;
  int p = blockIdx.x * 16 + (threadIdx.x >> 4);
  int lane = threadIdx.x & 15;
  if (p >= N) return;
  int b = p >> 12, hw = p & 4095, h = hw >> 6, w = hw & 63;

  float wt[35];
  #pragma unroll
  for (int k = 0; k < 35; ++k)
    wt[k] = sraw[(size_t)k * N + p] * sc[k] + sh[k];

  float m = wt[0];
  #pragma unroll
  for (int k = 1; k < 25; ++k) m = fmaxf(m, wt[k]);
  float s = 0.f;
  #pragma unroll
  for (int k = 0; k < 25; ++k) { wt[k] = __expf(wt[k] - m); s += wt[k]; }
  s = 1.f / s;
  #pragma unroll
  for (int k = 0; k < 25; ++k) wt[k] *= s;

  m = wt[25];
  #pragma unroll
  for (int k = 26; k < 34; ++k) m = fmaxf(m, wt[k]);
  s = 0.f;
  #pragma unroll
  for (int k = 25; k < 34; ++k) { wt[k] = __expf(wt[k] - m); s += wt[k]; }
  s = 1.f / s;
  #pragma unroll
  for (int k = 25; k < 34; ++k) wt[k] *= s;
  const float gate = wt[34];

  for (int c = lane; c < 256; c += 16) {
    const float* xch = x5 + ((size_t)(b * 256 + c) << 12);
    float a5 = 0.f; int k = 0;
    #pragma unroll
    for (int dy = -2; dy <= 2; ++dy) {
      int hh = h + dy; bool vy = (hh >= 0) & (hh < 64);
      #pragma unroll
      for (int dx = -2; dx <= 2; ++dx, ++k) {
        int ww = w + dx;
        if (vy && ww >= 0 && ww < 64) a5 += wt[k] * xch[hh * 64 + ww];
      }
    }
    float a3 = 0.f; k = 25;
    #pragma unroll
    for (int dy = -1; dy <= 1; ++dy) {
      int hh = h + dy; bool vy = (hh >= 0) & (hh < 64);
      #pragma unroll
      for (int dx = -1; dx <= 1; ++dx, ++k) {
        int ww = w + dx;
        if (vy && ww >= 0 && ww < 64) a3 += wt[k] * xch[hh * 64 + ww];
      }
    }
    float ctr = xch[hw];
    xc[(size_t)c * N + p]         = a5;
    xc[(size_t)(256 + c) * N + p] = a3;
    xc[(size_t)(512 + c) * N + p] = gate * ctr;
  }
}

// final: d_out (NCHW fp32) = relu(bn_l2(y2)), layout change matrix->NCHW
__global__ void final_out(const float* __restrict__ y, const float* sc,
                          const float* sh, float* __restrict__ out, int N) {
  int total = 256 * N;
  for (int i = blockIdx.x * blockDim.x + threadIdx.x; i < total;
       i += gridDim.x * blockDim.x) {
    int c = i / N, n = i - c * N;
    int b = n >> 12, hw = n & 4095;
    float v = fmaxf(y[i] * sc[c] + sh[c], 0.f);
    out[((size_t)(b * 256 + c) << 12) + hw] = v;
  }
}

// ---------------------------------------------------------------------------
// Host orchestration
// ---------------------------------------------------------------------------
template <int KS, int STRIDE, int PAD, int CI, int HI, int WI, int HO, int WO>
static void conv_call(hipStream_t st, const float* in, long long strC, long long strB,
                      const float* as, const float* ah, int relu,
                      const uint16_t* wp, const float* bias, float* out,
                      int Co, int Bn) {
  int Nout = Bn * HO * WO;
  dim3 grid(Nout / 64, (Co + 63) / 64);
  conv_wmma<KS, STRIDE, PAD, CI, HI, WI, HO, WO>
      <<<grid, 128, 0, st>>>(in, strC, strB, as, ah, relu, wp, bias, out, Co, Bn);
}

extern "C" void kernel_launch(void* const* d_in, const int* in_sizes, int n_in,
                              void* d_out, int out_size, void* d_ws, size_t ws_size,
                              hipStream_t stream) {
  (void)in_sizes; (void)n_in; (void)out_size; (void)ws_size;
  const float* x3 = (const float*)d_in[0];
  const float* x4 = (const float*)d_in[1];
  const float* x5 = (const float*)d_in[2];
  const float* P[43];
  for (int i = 3; i < 43; ++i) P[i] = (const float*)d_in[i];

  char* ws = (char*)d_ws;
  const size_t MiB = 1ull << 20;

  // packed bf16 weights
  uint16_t* WP1  = (uint16_t*)(ws + 0);        // 128x576
  uint16_t* WP2  = (uint16_t*)(ws + 147456);   // 256x1152
  uint16_t* WP3  = (uint16_t*)(ws + 737280);   // 256x1152
  uint16_t* WPF1 = (uint16_t*)(ws + 1327104);  // 256x256
  uint16_t* WPF2 = (uint16_t*)(ws + 1458176);  // 256x256
  uint16_t* WPK  = (uint16_t*)(ws + 1589248);  // 35x2304
  uint16_t* WPL1 = (uint16_t*)(ws + 1750528);  // 256x768
  uint16_t* WPL2 = (uint16_t*)(ws + 2143744);  // 256x256

  float* SUM = (float*)(ws + 4 * MiB);
  float* SSQ = (float*)(ws + 4 * MiB + 4096);
  char*  sb  = ws + 4 * MiB + 16384;
  float *SC1=(float*)(sb+0),     *SH1=(float*)(sb+1024);
  float *SC2=(float*)(sb+2048),  *SH2=(float*)(sb+3072);
  float *SC3=(float*)(sb+4096),  *SH3=(float*)(sb+5120);
  float *SCF1=(float*)(sb+6144), *SHF1=(float*)(sb+7168);
  float *SCF2=(float*)(sb+8192), *SHF2=(float*)(sb+9216);
  float *SCS=(float*)(sb+10240), *SHS=(float*)(sb+11264);
  float *SCL1=(float*)(sb+12288),*SHL1=(float*)(sb+13312);
  float *SCL2=(float*)(sb+14336),*SHL2=(float*)(sb+15360);
  float *BIAS35=(float*)(sb+16384), *G35=(float*)(sb+16640), *BE35=(float*)(sb+16896);

  // big buffers (with reuse)
  float* d1raw = (float*)(ws +  8 * MiB);  // [128][65536] 32 MiB
  float* d2raw = (float*)(ws + 40 * MiB);  // [256][16384] 16 MiB
  float* d3raw = (float*)(ws + 56 * MiB);  // [256][16384] 16 MiB
  float* x34   = (float*)(ws + 72 * MiB);  // [256][16384] 16 MiB
  float* xc    = (float*)(ws + 88 * MiB);  // [768][16384] 48 MiB
  float* f1    = (float*)(ws +  8 * MiB);  // reuse d1raw region
  float* f2    = (float*)(ws + 24 * MiB);  // reuse d1raw region (2nd half)
  float* sraw  = (float*)(ws + 40 * MiB);  // reuse d2raw
  float* y1    = (float*)(ws + 56 * MiB);  // reuse d3raw
  float* y2    = (float*)(ws + 72 * MiB);  // reuse x34

  // ---- weight packing ----
  pack_w3x3<<<288, 256, 0, stream>>>(P[3],  WP1, 128, 64);
  pack_w3x3<<<576, 256, 0, stream>>>(P[7],  WP2, 256, 128);
  pack_w3x3<<<576, 256, 0, stream>>>(P[11], WP3, 256, 128);
  pack_w3x3<<<226, 256, 0, stream>>>(P[23], WPK,              25, 256);
  pack_w3x3<<<81,  256, 0, stream>>>(P[27], WPK + 25 * 2304,   9, 256);
  pack_w3x3<<<9,   256, 0, stream>>>(P[31], WPK + 34 * 2304,   1, 256);
  pack_w1x1<<<256, 256, 0, stream>>>(P[15], WPF1, 256 * 256);
  pack_w1x1<<<256, 256, 0, stream>>>(P[19], WPF2, 256 * 256);
  pack_w1x1<<<768, 256, 0, stream>>>(P[35], WPL1, 256 * 768);
  pack_w1x1<<<256, 256, 0, stream>>>(P[39], WPL2, 256 * 256);
  concat35<<<1, 64, 0, stream>>>(P[24], P[28], P[32], BIAS35);
  concat35<<<1, 64, 0, stream>>>(P[25], P[29], P[33], G35);
  concat35<<<1, 64, 0, stream>>>(P[26], P[30], P[34], BE35);

  // ---- down1 conv a: x3 (NCHW 4x64x256x256) -> d1raw [128][65536] ----
  conv_call<3, 2, 1, 64, 256, 256, 128, 128>(
      stream, x3, 65536LL, 4194304LL, nullptr, nullptr, 0, WP1, P[4], d1raw, 128, 4);
  bn_stats<<<128, 256, 0, stream>>>(d1raw, SUM, SSQ, 65536);
  bn_finalize<<<1, 256, 0, stream>>>(SUM, SSQ, P[5], P[6], SC1, SH1, 128, 1.f / 65536.f);

  // ---- down1 conv b: bn1+relu folded into load -> d2raw [256][16384] ----
  conv_call<3, 2, 1, 128, 128, 128, 64, 64>(
      stream, d1raw, 65536LL, 16384LL, SC1, SH1, 1, WP2, P[8], d2raw, 256, 4);
  bn_stats<<<256, 256, 0, stream>>>(d2raw, SUM, SSQ, 16384);
  bn_finalize<<<1, 256, 0, stream>>>(SUM, SSQ, P[9], P[10], SC2, SH2, 256, 1.f / 16384.f);

  // ---- down2: x4 (NCHW 4x128x128x128) -> d3raw [256][16384] ----
  conv_call<3, 2, 1, 128, 128, 128, 64, 64>(
      stream, x4, 16384LL, 2097152LL, nullptr, nullptr, 0, WP3, P[12], d3raw, 256, 4);
  bn_stats<<<256, 256, 0, stream>>>(d3raw, SUM, SSQ, 16384);
  bn_finalize<<<1, 256, 0, stream>>>(SUM, SSQ, P[13], P[14], SC3, SH3, 256, 1.f / 16384.f);

  // ---- x34 = relu(bn2(d2raw) + bn3(d3raw)) ----
  x34_fuse<<<4096, 256, 0, stream>>>(d2raw, d3raw, SC2, SH2, SC3, SH3, x34, 16384);

  // ---- convFirst: two 1x1 GEMMs ----
  conv_call<1, 1, 0, 256, 64, 64, 64, 64>(
      stream, x34, 16384LL, 4096LL, nullptr, nullptr, 0, WPF1, P[16], f1, 256, 4);
  bn_stats<<<256, 256, 0, stream>>>(f1, SUM, SSQ, 16384);
  bn_finalize<<<1, 256, 0, stream>>>(SUM, SSQ, P[17], P[18], SCF1, SHF1, 256, 1.f / 16384.f);

  conv_call<1, 1, 0, 256, 64, 64, 64, 64>(
      stream, f1, 16384LL, 4096LL, SCF1, SHF1, 1, WPF2, P[20], f2, 256, 4);
  bn_stats<<<256, 256, 0, stream>>>(f2, SUM, SSQ, 16384);
  bn_finalize<<<1, 256, 0, stream>>>(SUM, SSQ, P[21], P[22], SCF2, SHF2, 256, 1.f / 16384.f);

  // ---- fused kernel-prediction conv: 256 -> 35 ch, 3x3 s1 p1 ----
  conv_call<3, 1, 1, 256, 64, 64, 64, 64>(
      stream, f2, 16384LL, 4096LL, SCF2, SHF2, 1, WPK, BIAS35, sraw, 35, 4);
  bn_stats<<<35, 256, 0, stream>>>(sraw, SUM, SSQ, 16384);
  bn_finalize<<<1, 64, 0, stream>>>(SUM, SSQ, G35, BE35, SCS, SHS, 35, 1.f / 16384.f);

  // ---- dynamic filtering -> xc [768][16384] ----
  dynfilter<<<1024, 256, 0, stream>>>(sraw, SCS, SHS, x5, xc, 4);

  // ---- convLast: 3C->C, then C->C ----
  conv_call<1, 1, 0, 768, 64, 64, 64, 64>(
      stream, xc, 16384LL, 4096LL, nullptr, nullptr, 0, WPL1, P[36], y1, 256, 4);
  bn_stats<<<256, 256, 0, stream>>>(y1, SUM, SSQ, 16384);
  bn_finalize<<<1, 256, 0, stream>>>(SUM, SSQ, P[37], P[38], SCL1, SHL1, 256, 1.f / 16384.f);

  conv_call<1, 1, 0, 256, 64, 64, 64, 64>(
      stream, y1, 16384LL, 4096LL, SCL1, SHL1, 1, WPL2, P[40], y2, 256, 4);
  bn_stats<<<256, 256, 0, stream>>>(y2, SUM, SSQ, 16384);
  bn_finalize<<<1, 256, 0, stream>>>(SUM, SSQ, P[41], P[42], SCL2, SHL2, 256, 1.f / 16384.f);

  // ---- final affine + relu, matrix -> NCHW fp32 ----
  final_out<<<4096, 256, 0, stream>>>(y2, SCL2, SHL2, (float*)d_out, 16384);
}
